// moegpt_26276609917411
// MI455X (gfx1250) — compile-verified, matrix operand order checked by reading, not run on hardware
//
#include <hip/hip_runtime.h>

// ---------------------------------------------------------------------------
// CDNA5 (gfx1250) WMMA types
// ---------------------------------------------------------------------------
typedef __attribute__((ext_vector_type(16))) __bf16          v16bf;
typedef __attribute__((ext_vector_type(16))) unsigned short  v16u;
typedef __attribute__((ext_vector_type(8)))  float           v8f;
typedef int v4i_t __attribute__((vector_size(16)));   // matches builtin proto

#define S_LEN   4096
#define NHEADS  16
#define WIN_SZ  1024
#define DOC_TOK 50256

#define AS1 __attribute__((address_space(1)))
#define AS3 __attribute__((address_space(3)))

#if __has_builtin(__builtin_amdgcn_global_load_async_to_lds_b128)
#define HAVE_ASYNC_LDS 1
#else
#define HAVE_ASYNC_LDS 0
#endif

__device__ __forceinline__ void wait_async_lds() {
#if HAVE_ASYNC_LDS && __has_builtin(__builtin_amdgcn_s_wait_asynccnt)
    __builtin_amdgcn_s_wait_asynccnt(0);
#endif
}

#if HAVE_ASYNC_LDS
// async DMA of one 16B chunk: global (AS1) -> LDS (AS3), tracked by ASYNCcnt
__device__ __forceinline__ void async_lds_b128(const void* gsrc, void* ldst) {
    __builtin_amdgcn_global_load_async_to_lds_b128(
        (AS1 v4i_t*)gsrc, (AS3 v4i_t*)ldst, 0, 0);
}
#endif

// Native CDNA5 bf16 convert (RTNE), single VALU op instead of bit twiddling.
__device__ __forceinline__ unsigned short f32_to_bf16_raw(float f) {
    return __builtin_bit_cast(unsigned short, (__bf16)f);
}

// k index of the first element of VGPR pair v for the 16-bit A/B operand layout
// (CDNA5 ISA 7.12.2: lanes 0-15 hold K 0-7 & 16-23, lanes 16-31 hold K 8-15 & 24-31)
__device__ __forceinline__ int kpair(int v, int khalf) {
    return ((v < 4) ? 0 : 16) + khalf * 8 + (v & 3) * 2;
}

__device__ __forceinline__ v8f wmma_bf16(const v16u& a, const v16u& b, v8f c) {
    return __builtin_amdgcn_wmma_f32_16x16x32_bf16(
        false, __builtin_bit_cast(v16bf, a),
        false, __builtin_bit_cast(v16bf, b),
        (short)0, c, false, false);
}

// ---------------------------------------------------------------------------
// Generic GEMM:  C[M,N] = A[M,K] @ B[K,N] + bias   (bf16 WMMA, f32 accumulate)
// Block tile 128x64, 256 threads (8 waves), K step 32.
// ABF: A is raw-bf16 (else f32).  OBF: output raw-bf16 (else f32).
// Requires M%128==0, N%64==0, K%32==0.
// ---------------------------------------------------------------------------
template<int ABF, int OBF>
__global__ __launch_bounds__(256) void gemm_wmma(
    const void* __restrict__ Ap, const float* __restrict__ B,
    const float* __restrict__ bias, void* __restrict__ Cp,
    int M, int N, int K)
{
    __shared__ unsigned short a_lds[128 * 32];
    __shared__ unsigned short b_lds[64 * 32];   // stored transposed: [n][k]

    const int tid   = threadIdx.x;
    const int wave  = tid >> 5;
    const int lane  = tid & 31;
    const int lm    = lane & 15;
    const int khalf = lane >> 4;
    const int bm0   = blockIdx.x * 128;
    const int bn0   = blockIdx.y * 64;

    v8f acc[4];
#pragma unroll
    for (int nt = 0; nt < 4; ++nt)
#pragma unroll
        for (int i = 0; i < 8; ++i) acc[nt][i] = 0.0f;

    for (int k0 = 0; k0 < K; k0 += 32) {
        __syncthreads();
        // ---- stage A tile (128 x 32) into LDS as bf16 ----
        if (ABF) {
#if HAVE_ASYNC_LDS
            // async DMA: 512 x 16B chunks, 256 threads, 2 chunks each
#pragma unroll
            for (int it = 0; it < 2; ++it) {
                int c   = it * 256 + tid;
                int row = c >> 2;
                int col = (c & 3) * 8;
                const unsigned short* src =
                    (const unsigned short*)Ap + (size_t)(bm0 + row) * K + k0 + col;
                async_lds_b128(src, a_lds + c * 8);
            }
#else
            int row = tid >> 1;
            int c0  = (tid & 1) * 16;
            unsigned short* d = a_lds + row * 32 + c0;
            const unsigned short* A =
                (const unsigned short*)Ap + (size_t)(bm0 + row) * K + k0 + c0;
            ((uint4*)d)[0] = ((const uint4*)A)[0];
            ((uint4*)d)[1] = ((const uint4*)A)[1];
#endif
        } else {
            int row = tid >> 1;
            int c0  = (tid & 1) * 16;
            unsigned short* d = a_lds + row * 32 + c0;
            const float* A = (const float*)Ap + (size_t)(bm0 + row) * K + k0 + c0;
#pragma unroll
            for (int i = 0; i < 16; ++i) d[i] = f32_to_bf16_raw(A[i]);
        }
        // ---- stage B tile (32 x 64) transposed into LDS as bf16 ----
        {
            int row = tid >> 3;
            int c0  = (tid & 7) * 8;
            const float* Bp = B + (size_t)(k0 + row) * N + bn0 + c0;
#pragma unroll
            for (int i = 0; i < 8; ++i)
                b_lds[(c0 + i) * 32 + row] = f32_to_bf16_raw(Bp[i]);
        }
        if (ABF) wait_async_lds();
        __syncthreads();

        // ---- A operand fragment (wave's 16 rows) ----
        v16u au;
        const unsigned short* ar = a_lds + (wave * 16 + lm) * 32;
#pragma unroll
        for (int v = 0; v < 8; ++v) {
            unsigned int w = *(const unsigned int*)(ar + kpair(v, khalf));
            au[2 * v]     = (unsigned short)(w & 0xffffu);
            au[2 * v + 1] = (unsigned short)(w >> 16);
        }
        // ---- 4 WMMAs across the 64-wide N tile ----
#pragma unroll
        for (int nt = 0; nt < 4; ++nt) {
            v16u bu;
            const unsigned short* br = b_lds + (nt * 16 + lm) * 32;
#pragma unroll
            for (int v = 0; v < 8; ++v) {
                unsigned int w = *(const unsigned int*)(br + kpair(v, khalf));
                bu[2 * v]     = (unsigned short)(w & 0xffffu);
                bu[2 * v + 1] = (unsigned short)(w >> 16);
            }
            acc[nt] = wmma_bf16(au, bu, acc[nt]);
        }
    }

    // ---- epilogue: bias + store (C/D layout: m = r + 8*khalf, n = lm) ----
#pragma unroll
    for (int nt = 0; nt < 4; ++nt) {
        int col  = bn0 + nt * 16 + lm;
        float bv = bias ? bias[col] : 0.0f;
#pragma unroll
        for (int r = 0; r < 8; ++r) {
            int row = bm0 + wave * 16 + r + 8 * khalf;
            float v = acc[nt][r] + bv;
            if (OBF)
                ((unsigned short*)Cp)[(size_t)row * N + col] = f32_to_bf16_raw(v);
            else
                ((float*)Cp)[(size_t)row * N + col] = v;
        }
    }
}

// ---------------------------------------------------------------------------
// RoPE: in f32 [S,64] -> out raw-bf16 [S,64]
// ---------------------------------------------------------------------------
__global__ void rope_kernel(const float* __restrict__ in,
                            unsigned short* __restrict__ out, int S)
{
    int idx = blockIdx.x * blockDim.x + threadIdx.x;
    if (idx >= S * 32) return;
    int s = idx >> 5, j = idx & 31;
    float inv = __expf((float)(2 * j) * (-9.210340371976184f / 64.0f));
    float ang = (float)s * inv;
    float sn, cs;
    __sincosf(ang, &sn, &cs);
    float a = in[s * 64 + j];
    float b = in[s * 64 + j + 32];
    out[s * 64 + j]      = f32_to_bf16_raw(a * cs - b * sn);
    out[s * 64 + j + 32] = f32_to_bf16_raw(b * cs + a * sn);
}

// ---------------------------------------------------------------------------
// Document-id cumsum (tiny, serial)
// ---------------------------------------------------------------------------
__global__ void docs_kernel(const int* __restrict__ tok, int* __restrict__ docs, int S)
{
    if (blockIdx.x == 0 && threadIdx.x == 0) {
        int acc = 0;
        for (int i = 0; i < S; ++i) {
            acc += (tok[i] == DOC_TOK) ? 1 : 0;
            docs[i] = acc;
        }
    }
}

// ---------------------------------------------------------------------------
// Flash attention, sliding-window + doc mask.
// grid = (S/64, NHEADS), block = 128 (4 waves). Each wave owns 16 query rows.
// q_bf [S,1024] head-major 64 + qr_bf [S,64] form the 128-dim q_full.
// ---------------------------------------------------------------------------
__global__ __launch_bounds__(128) void attn_kernel(
    const unsigned short* __restrict__ q_bf,
    const unsigned short* __restrict__ qr_bf,
    const unsigned short* __restrict__ k_bf,
    const unsigned short* __restrict__ kr_bf,
    const unsigned short* __restrict__ v_bf,
    const int* __restrict__ docs,
    unsigned short* __restrict__ o_bf)
{
    __shared__ unsigned short k_lds[64 * 128];    // [key][dim]
    __shared__ unsigned short vt_lds[128 * 64];   // [dim][key]  (transposed V)
    __shared__ unsigned short p_lds[64 * 64];     // [row][key]

    const int h     = blockIdx.y;
    const int qt    = blockIdx.x;
    const int q0    = qt * 64;
    const int tid   = threadIdx.x;
    const int wave  = tid >> 5;
    const int lane  = tid & 31;
    const int lm    = lane & 15;
    const int khalf = lane >> 4;
    const float SCALE = 0.08838834764831845f;     // (2*64)^-0.5

    // ---- Q fragments (16 rows x 128 dims = 4 K-chunks), kept in VGPRs ----
    v16u qf[4];
    {
        const int qrow = q0 + wave * 16 + lm;
#pragma unroll
        for (int kc = 0; kc < 4; ++kc) {
            const unsigned short* src = (kc < 2)
                ? (q_bf + (size_t)qrow * 1024 + h * 64 + kc * 32)
                : (qr_bf + (size_t)qrow * 64 + (kc - 2) * 32);
#pragma unroll
            for (int v = 0; v < 8; ++v) {
                unsigned int w = *(const unsigned int*)(src + kpair(v, khalf));
                qf[kc][2 * v]     = (unsigned short)(w & 0xffffu);
                qf[kc][2 * v + 1] = (unsigned short)(w >> 16);
            }
        }
    }

    int docq[8];
#pragma unroll
    for (int r = 0; r < 8; ++r) docq[r] = docs[q0 + wave * 16 + r + 8 * khalf];

    float m_st[8], l_st[8];
    v8f o_acc[8];
#pragma unroll
    for (int r = 0; r < 8; ++r) { m_st[r] = -3e38f; l_st[r] = 0.0f; }
#pragma unroll
    for (int t = 0; t < 8; ++t)
#pragma unroll
        for (int i = 0; i < 8; ++i) o_acc[t][i] = 0.0f;

    int kt0 = qt - 16; if (kt0 < 0) kt0 = 0;

    for (int kt = kt0; kt <= qt; ++kt) {
        const int kbase = kt * 64;
        __syncthreads();   // protect LDS reuse across iterations

        // ---- stage K tile (64 keys x 128 dims) ----
#if HAVE_ASYNC_LDS
        {
            // async DMA straight into LDS: 1024 x 16B chunks / 128 threads
#pragma unroll
            for (int it = 0; it < 8; ++it) {
                int c    = it * 128 + tid;
                int row  = c >> 4;          // 16 chunks per 128-dim row
                int col  = (c & 15) * 8;
                int srow = kbase + row;
                const unsigned short* src = (col < 64)
                    ? (k_bf + (size_t)srow * 1024 + h * 64 + col)
                    : (kr_bf + (size_t)srow * 64 + (col - 64));
                async_lds_b128(src, k_lds + c * 8);
            }
        }
#else
        {
            int r = tid >> 1, half = tid & 1;
            int srow = kbase + r;
            const unsigned short* src = half
                ? (kr_bf + (size_t)srow * 64)
                : (k_bf + (size_t)srow * 1024 + h * 64);
            unsigned short* dst = k_lds + r * 128 + half * 64;
#pragma unroll
            for (int i = 0; i < 8; ++i) ((uint4*)dst)[i] = ((const uint4*)src)[i];
        }
#endif
        // ---- stage V tile transposed (dim-major) ----
        {
            int r = tid >> 1, half = tid & 1;
            int srow = kbase + r;
            const unsigned int* s32 =
                (const unsigned int*)(v_bf + (size_t)srow * 2048 + h * 128 + half * 64);
#pragma unroll
            for (int i = 0; i < 32; ++i) {
                unsigned int w = s32[i];
                vt_lds[(half * 64 + 2 * i)     * 64 + r] = (unsigned short)(w & 0xffffu);
                vt_lds[(half * 64 + 2 * i + 1) * 64 + r] = (unsigned short)(w >> 16);
            }
        }
        // prefetch next K tile while we compute (global_prefetch_b8)
        if (kt < qt) {
            __builtin_prefetch(k_bf + (size_t)(kbase + 64) * 1024 + h * 64, 0, 1);
            __builtin_prefetch(v_bf + (size_t)(kbase + 64) * 2048 + h * 128, 0, 1);
        }
        wait_async_lds();
        __syncthreads();

        // ---- S = Q * K^T : 4 n-subtiles x 4 K-chunks = 16 WMMAs ----
        v8f sacc[4];
#pragma unroll
        for (int ns = 0; ns < 4; ++ns) {
            v8f a;
#pragma unroll
            for (int i = 0; i < 8; ++i) a[i] = 0.0f;
#pragma unroll
            for (int kc = 0; kc < 4; ++kc) {
                v16u bu;
                const unsigned short* krow = k_lds + (ns * 16 + lm) * 128 + kc * 32;
#pragma unroll
                for (int v = 0; v < 8; ++v) {
                    unsigned int w = *(const unsigned int*)(krow + kpair(v, khalf));
                    bu[2 * v]     = (unsigned short)(w & 0xffffu);
                    bu[2 * v + 1] = (unsigned short)(w >> 16);
                }
                a = wmma_bf16(qf[kc], bu, a);
            }
            sacc[ns] = a;
        }

        // ---- mask + online softmax ----
        int kpos[4], dock[4];
#pragma unroll
        for (int ns = 0; ns < 4; ++ns) {
            kpos[ns] = kbase + ns * 16 + lm;
            dock[ns] = docs[kpos[ns]];
        }
#pragma unroll
        for (int r = 0; r < 8; ++r) {
            int qpos = q0 + wave * 16 + r + 8 * khalf;
            float mx = -3e38f;
#pragma unroll
            for (int ns = 0; ns < 4; ++ns) {
                int rel = qpos - kpos[ns];
                bool ok = (rel >= 0) && (rel < WIN_SZ) && (docq[r] == dock[ns]);
                float sv = ok ? sacc[ns][r] * SCALE : -3e38f;
                sacc[ns][r] = sv;
                mx = fmaxf(mx, sv);
            }
#pragma unroll
            for (int off = 1; off < 16; off <<= 1)
                mx = fmaxf(mx, __shfl_xor(mx, off, 32));

            float mold  = m_st[r];
            float mnew  = fmaxf(mold, mx);
            float alpha = __expf(mold - mnew);
            float rs = 0.0f;
#pragma unroll
            for (int ns = 0; ns < 4; ++ns) {
                float sv = sacc[ns][r];
                float p  = (sv > -1e37f) ? __expf(sv - mnew) : 0.0f;
                sacc[ns][r] = p;
                rs += p;
            }
#pragma unroll
            for (int off = 1; off < 16; off <<= 1)
                rs += __shfl_xor(rs, off, 32);

            l_st[r] = l_st[r] * alpha + rs;
            m_st[r] = mnew;
#pragma unroll
            for (int t = 0; t < 8; ++t) o_acc[t][r] *= alpha;
        }

        // ---- P through LDS (C/D layout -> A-operand layout) ----
#pragma unroll
        for (int ns = 0; ns < 4; ++ns)
#pragma unroll
            for (int r = 0; r < 8; ++r)
                p_lds[(wave * 16 + r + 8 * khalf) * 64 + ns * 16 + lm] =
                    f32_to_bf16_raw(sacc[ns][r]);
        __syncthreads();

        // ---- O += P * V : 2 K-chunks x 8 n-tiles = 16 WMMAs ----
#pragma unroll
        for (int kc = 0; kc < 2; ++kc) {
            v16u pu;
            const unsigned short* prow = p_lds + (wave * 16 + lm) * 64 + kc * 32;
#pragma unroll
            for (int v = 0; v < 8; ++v) {
                unsigned int w = *(const unsigned int*)(prow + kpair(v, khalf));
                pu[2 * v]     = (unsigned short)(w & 0xffffu);
                pu[2 * v + 1] = (unsigned short)(w >> 16);
            }
#pragma unroll
            for (int nt = 0; nt < 8; ++nt) {
                v16u bu;
                const unsigned short* vrow = vt_lds + (nt * 16 + lm) * 64 + kc * 32;
#pragma unroll
                for (int v = 0; v < 8; ++v) {
                    unsigned int w = *(const unsigned int*)(vrow + kpair(v, khalf));
                    bu[2 * v]     = (unsigned short)(w & 0xffffu);
                    bu[2 * v + 1] = (unsigned short)(w >> 16);
                }
                o_acc[nt] = wmma_bf16(pu, bu, o_acc[nt]);
            }
        }
    }

    // ---- normalize + store (row-major [S, NH*128], head-major columns) ----
#pragma unroll
    for (int r = 0; r < 8; ++r) {
        float inv = 1.0f / l_st[r];
        int srow = q0 + wave * 16 + r + 8 * khalf;
        unsigned short* dst = o_bf + (size_t)srow * 2048 + h * 128;
#pragma unroll
        for (int nt = 0; nt < 8; ++nt)
            dst[nt * 16 + lm] = f32_to_bf16_raw(o_acc[nt][r] * inv);
    }
}

// ---------------------------------------------------------------------------
// Orchestration
// ---------------------------------------------------------------------------
extern "C" void kernel_launch(void* const* d_in, const int* in_sizes, int n_in,
                              void* d_out, int out_size, void* d_ws, size_t ws_size,
                              hipStream_t stream)
{
    (void)in_sizes; (void)n_in; (void)out_size; (void)ws_size;

    const float* x    = (const float*)d_in[0];
    const float* Wqd  = (const float*)d_in[1];
    const float* bqd  = (const float*)d_in[2];
    const float* Wqu  = (const float*)d_in[3];
    const float* bqu  = (const float*)d_in[4];
    const float* Wqr  = (const float*)d_in[5];
    const float* bqr  = (const float*)d_in[6];
    const float* Wkvd = (const float*)d_in[7];
    const float* bkvd = (const float*)d_in[8];
    const float* Wku  = (const float*)d_in[9];
    const float* bku  = (const float*)d_in[10];
    const float* Wvu  = (const float*)d_in[11];
    const float* bvu  = (const float*)d_in[12];
    const float* Wkr  = (const float*)d_in[13];
    const float* bkr  = (const float*)d_in[14];
    const float* Wo   = (const float*)d_in[15];
    const float* bo   = (const float*)d_in[16];
    const int*   tok  = (const int*)d_in[17];
    float* out = (float*)d_out;

    char* ws = (char*)d_ws;
    size_t off = 0;
    auto alloc = [&](size_t bytes) -> void* {
        void* p = ws + off;
        off += (bytes + 255) & ~(size_t)255;
        return p;
    };
    unsigned short* qd_bf   = (unsigned short*)alloc((size_t)S_LEN * 512 * 2);
    unsigned short* kv_bf   = (unsigned short*)alloc((size_t)S_LEN * 512 * 2);
    float*          qr_pre  = (float*)alloc((size_t)S_LEN * 64 * 4);
    float*          kr_pre  = (float*)alloc((size_t)S_LEN * 64 * 4);
    unsigned short* qr_bf   = (unsigned short*)alloc((size_t)S_LEN * 64 * 2);
    unsigned short* kr_bf   = (unsigned short*)alloc((size_t)S_LEN * 64 * 2);
    unsigned short* q_bf    = (unsigned short*)alloc((size_t)S_LEN * 1024 * 2);
    unsigned short* k_bf    = (unsigned short*)alloc((size_t)S_LEN * 1024 * 2);
    unsigned short* v_bf    = (unsigned short*)alloc((size_t)S_LEN * 2048 * 2);
    unsigned short* o_bf    = (unsigned short*)alloc((size_t)S_LEN * 2048 * 2);
    int*            docs    = (int*)alloc((size_t)S_LEN * 4);

    const dim3 blk(256);
    auto grid_of = [](int M, int N) { return dim3(M / 128, N / 64); };

    // doc ids
    docs_kernel<<<1, 1, 0, stream>>>(tok, docs, S_LEN);

    // down-projections + rope pre-projections (A = f32 x)
    gemm_wmma<0, 1><<<grid_of(S_LEN, 512), blk, 0, stream>>>(x, Wqd,  bqd,  qd_bf,  S_LEN, 512,  1024);
    gemm_wmma<0, 1><<<grid_of(S_LEN, 512), blk, 0, stream>>>(x, Wkvd, bkvd, kv_bf,  S_LEN, 512,  1024);
    gemm_wmma<0, 0><<<grid_of(S_LEN, 64),  blk, 0, stream>>>(x, Wqr,  bqr,  qr_pre, S_LEN, 64,   1024);
    gemm_wmma<0, 0><<<grid_of(S_LEN, 64),  blk, 0, stream>>>(x, Wkr,  bkr,  kr_pre, S_LEN, 64,   1024);

    // rope -> bf16
    rope_kernel<<<(S_LEN * 32 + 255) / 256, 256, 0, stream>>>(qr_pre, qr_bf, S_LEN);
    rope_kernel<<<(S_LEN * 32 + 255) / 256, 256, 0, stream>>>(kr_pre, kr_bf, S_LEN);

    // up-projections (A = bf16 intermediates)
    gemm_wmma<1, 1><<<grid_of(S_LEN, 1024), blk, 0, stream>>>(qd_bf, Wqu, bqu, q_bf, S_LEN, 1024, 512);
    gemm_wmma<1, 1><<<grid_of(S_LEN, 1024), blk, 0, stream>>>(kv_bf, Wku, bku, k_bf, S_LEN, 1024, 512);
    gemm_wmma<1, 1><<<grid_of(S_LEN, 2048), blk, 0, stream>>>(kv_bf, Wvu, bvu, v_bf, S_LEN, 2048, 512);

    // attention
    attn_kernel<<<dim3(S_LEN / 64, NHEADS), dim3(128), 0, stream>>>(
        q_bf, qr_bf, k_bf, kr_bf, v_bf, docs, o_bf);

    // output projection (f32 out)
    gemm_wmma<1, 0><<<grid_of(S_LEN, 1024), blk, 0, stream>>>(o_bf, Wo, bo, out, S_LEN, 1024, 2048);
}
